// HybridHyenaGolfModel_47519518163604
// MI455X (gfx1250) — matrix operand order repaired and checked by reading, not compiled, not run on hardware
//
#include <hip/hip_runtime.h>

typedef _Float16 f16;
typedef _Float16 v16h __attribute__((ext_vector_type(16)));
typedef float    v8f  __attribute__((ext_vector_type(8)));

#define WMMA16(a,b,c) __builtin_amdgcn_wmma_f32_16x16x32_f16(false,(a),false,(b),(short)0,(c),false,false)

static constexpr int T_ = 1024;
static constexpr int D_ = 1024;
static constexpr int H_ = 8;

// ---- CDNA5 async global->LDS copy (ASYNCcnt path, no VGPR round-trip) ----
// LDS operand is the low 32 bits of the generic pointer (flat->LDS truncation
// yields the wave-relative LDS byte address).
__device__ __forceinline__ void async_copy16(void* lds_ptr, const void* gptr){
  uint32_t lds = (uint32_t)(uintptr_t)lds_ptr;
  uint64_t ga  = (uint64_t)(uintptr_t)gptr;
  asm volatile("global_load_async_to_lds_b128 %0, %1, off"
               :: "v"(lds), "v"(ga) : "memory");
}
// Async loads complete in-order, so waiting for "<= N outstanding" retires all
// but the most recently issued N copies (the next pipeline stage).
__device__ __forceinline__ void wait_async0(){  asm volatile("s_wait_asynccnt 0x0"  ::: "memory"); }
__device__ __forceinline__ void wait_async3(){  asm volatile("s_wait_asynccnt 0x3"  ::: "memory"); }
__device__ __forceinline__ void wait_async32(){ asm volatile("s_wait_asynccnt 0x20" ::: "memory"); }

// ---------------- block reductions (256 threads = 8 wave32) ----------------
__device__ __forceinline__ float blk_sum256(float v){
  #pragma unroll
  for (int off = 16; off; off >>= 1) v += __shfl_xor(v, off, 32);
  __shared__ float red[8];
  if ((threadIdx.x & 31) == 0) red[threadIdx.x >> 5] = v;
  __syncthreads();
  if (threadIdx.x == 0){ float s = 0.f; for (int i = 0; i < 8; i++) s += red[i]; red[0] = s; }
  __syncthreads();
  float r = red[0];
  __syncthreads();
  return r;
}
__device__ __forceinline__ float blk_max256(float v){
  #pragma unroll
  for (int off = 16; off; off >>= 1) v = fmaxf(v, __shfl_xor(v, off, 32));
  __shared__ float red[8];
  if ((threadIdx.x & 31) == 0) red[threadIdx.x >> 5] = v;
  __syncthreads();
  if (threadIdx.x == 0){ float s = red[0]; for (int i = 1; i < 8; i++) s = fmaxf(s, red[i]); red[0] = s; }
  __syncthreads();
  float r = red[0];
  __syncthreads();
  return r;
}

// ---------------- elementwise / prep kernels ----------------
__global__ __launch_bounds__(256) void k_embed(const int* __restrict__ ids,
                                               const float* __restrict__ emb,
                                               float* __restrict__ x){
  int tok = blockIdx.x;
  int id  = ids[tok];
  const float* src = emb + (size_t)id * D_;
  float* dst = x + (size_t)tok * D_;
  for (int c = threadIdx.x; c < D_; c += 256) dst[c] = src[c];
}

__global__ __launch_bounds__(256) void k_rmsnorm(const float* __restrict__ x,
                                                 const float* __restrict__ w,
                                                 f16* __restrict__ o){
  int row = blockIdx.x;
  const float* xr = x + (size_t)row * D_;
  float ss = 0.f;
  for (int c = threadIdx.x; c < D_; c += 256){ float v = xr[c]; ss += v*v; }
  float tot = blk_sum256(ss);
  float rs  = rsqrtf(tot * (1.0f / D_) + 1e-6f);
  f16* orow = o + (size_t)row * D_;
  for (int c = threadIdx.x; c < D_; c += 256) orow[c] = (f16)(xr[c] * rs * w[c]);
}

// per-row fake-quant (scale = max|row|/127, round-half-even) then f16
__global__ __launch_bounds__(256) void k_fq(const float* __restrict__ w,
                                            f16* __restrict__ o, int K){
  int row = blockIdx.x;
  const float* wr = w + (size_t)row * K;
  float mx = 0.f;
  for (int c = threadIdx.x; c < K; c += 256) mx = fmaxf(mx, fabsf(wr[c]));
  float m = blk_max256(mx);
  float scale = fmaxf(m * (1.0f / 127.0f), 1e-8f);
  float inv = 1.0f / scale;
  f16* orow = o + (size_t)row * K;
  for (int c = threadIdx.x; c < K; c += 256) orow[c] = (f16)(rintf(wr[c] * inv) * scale);
}

__global__ __launch_bounds__(256) void k_cvt(const float* __restrict__ a,
                                             f16* __restrict__ o, size_t n){
  size_t i = (size_t)blockIdx.x * 256 + threadIdx.x;
  size_t stride = (size_t)gridDim.x * 256;
  for (; i < n; i += stride) o[i] = (f16)a[i];
}

// Hyena "FFT conv" == 1st-order IIR (filt is exponential): y[t]=f0*x[t]+a*y[t-1].
// Fused with SiLU gate; one thread per (b,d) channel, coalesced over d.
__global__ __launch_bounds__(256) void k_scan(const float* __restrict__ qkv,
                                              const float* __restrict__ filt,
                                              f16* __restrict__ o){
  int gid = blockIdx.x * 256 + threadIdx.x;   // 0..2047 == b*D + d
  int d = gid & (D_ - 1);
  int b = gid >> 10;
  int hh = d >> 7;                             // DH = 128
  float f0 = filt[hh];
  float f1 = filt[H_ + hh];
  float a = (f0 != 0.f) ? f1 / f0 : 0.f;
  const float* q = qkv + (size_t)b * T_ * (2 * D_);
  f16* ob = o + (size_t)b * T_ * D_;
  float y = 0.f;
  for (int t = 0; t < T_; ++t){
    float s = q[(size_t)t * (2 * D_) + d];
    y = f0 * s + a * y;
    float g = q[(size_t)t * (2 * D_) + D_ + d];
    float si = g / (1.0f + __expf(-g));
    ob[(size_t)t * D_ + d] = (f16)(y * si);
  }
}

// bigram/trigram hash embeddings: x += bi_emb[h_bi] @ bi_proj^T + tri_emb[h_tri] @ tri_proj^T
__global__ __launch_bounds__(256) void k_hash(const int* __restrict__ ids,
    const float* __restrict__ bi_emb, const float* __restrict__ bi_proj,
    const float* __restrict__ tri_emb, const float* __restrict__ tri_proj,
    float* x){
  int tok = blockIdx.x;
  int t = tok & (T_ - 1);
  int id = ids[tok];
  int s1 = (t >= 1) ? ids[tok - 1] : 0;
  int s2 = (t >= 2) ? ids[tok - 2] : 0;
  int hb = (int)((unsigned)s1 * 31337u) ^ id;
  hb %= 20480; if (hb < 0) hb += 20480;
  int ht = ((int)((unsigned)s2 * 31337u)) ^ ((int)((unsigned)s1 * 179424673u)) ^ id;
  ht %= 10240; if (ht < 0) ht += 10240;
  __shared__ float eb[64], et[64];
  if (threadIdx.x < 64)       eb[threadIdx.x]      = bi_emb[(size_t)hb * 64 + threadIdx.x];
  else if (threadIdx.x < 128) et[threadIdx.x - 64] = tri_emb[(size_t)ht * 64 + (threadIdx.x - 64)];
  __syncthreads();
  float* xr = x + (size_t)tok * D_;
  for (int d = threadIdx.x; d < D_; d += 256){
    float acc = 0.f;
    const float* bp = bi_proj + (size_t)d * 64;
    const float* tp = tri_proj + (size_t)d * 64;
    #pragma unroll 8
    for (int e = 0; e < 64; ++e) acc += bp[e] * eb[e] + tp[e] * et[e];
    xr[d] += acc;
  }
}

// ---------------- WMMA GEMM: C[M,N] = A[M,K] * W[N,K]^T  (f16 in, f32 acc) ----------
// 256 threads = 8 waves; block tile 128x64; each wave owns 32x32 = 2x2 WMMA tiles.
// Double-buffered LDS staging via global_load_async_to_lds_b128: tile kt+1's
// fetch overlaps WMMA on tile kt; s_wait_asynccnt 3 retires everything but the
// 3 in-flight copies of the next tile (async completions are in-order).
// EP: 0 = f32 out (+optional residual), 1 = gelu->f16, 2 = f16 passthrough.
template<int EP>
__global__ __launch_bounds__(256) void k_gemm(const f16* __restrict__ A,
                                              const f16* __restrict__ W,
                                              const float* Res, float* C32,
                                              f16* __restrict__ C16,
                                              int M, int N, int K){
  __shared__ f16 As[2][128][40];   // +8 half pad, rows stay 16B aligned
  __shared__ f16 Bs[2][64][40];
  (void)M;
  const int m0 = blockIdx.y * 128;
  const int n0 = blockIdx.x * 64;
  const int tid  = threadIdx.x;
  const int lane = tid & 31, wi = tid >> 5;
  const int half = lane >> 4, lm = lane & 15;
  const int wm = wi >> 1, wn = wi & 1;
  const v8f vz = {0.f,0.f,0.f,0.f,0.f,0.f,0.f,0.f};
  v8f acc[2][2];
  acc[0][0] = vz; acc[0][1] = vz; acc[1][0] = vz; acc[1][1] = vz;
  // per-thread staging assignments (constant across K loop)
  const int ra0 = (tid * 2) >> 2,     sa0 = (tid * 2) & 3;
  const int ra1 = (tid * 2 + 1) >> 2, sa1 = (tid * 2 + 1) & 3;
  const int rb  = tid >> 2,           sb  = tid & 3;
  const int KT = K >> 5;
  auto issue = [&](int buf, int kt){
    const int kb = kt << 5;
    async_copy16(&As[buf][ra0][sa0 * 8], A + (size_t)(m0 + ra0) * K + kb + sa0 * 8);
    async_copy16(&As[buf][ra1][sa1 * 8], A + (size_t)(m0 + ra1) * K + kb + sa1 * 8);
    async_copy16(&Bs[buf][rb ][sb  * 8], W + (size_t)(n0 + rb ) * K + kb + sb  * 8);
  };
  issue(0, 0);
  if (KT > 1) issue(1, 1);
  for (int kt = 0; kt < KT; ++kt){
    if (kt + 2 < KT){                 // warm L2 two tiles ahead of the async engine
      __builtin_prefetch(A + (size_t)(m0 + (tid >> 1)) * K + ((kt + 2) << 5), 0, 0);
      __builtin_prefetch(W + (size_t)(n0 + (tid >> 2)) * K + ((kt + 2) << 5), 0, 0);
    }
    if (kt + 1 < KT) wait_async3(); else wait_async0();
    __syncthreads();
    const int buf = kt & 1;
    // Fragment packing per ISA 7.12.2 (wave32):
    //  A 16x32 f16: lane holds row M=lane%16, K = (e<8?e:e+8)+8*(lane/16)
    //  B 32x16 f16: lane holds col N=lane%16, K = (e<8?e:e+8)+8*(lane/16)
    v16h aF[2], bF[2];
    #pragma unroll
    for (int t = 0; t < 2; ++t){
      #pragma unroll
      for (int e = 0; e < 16; ++e){
        int ka = (e < 8 ? e : e + 8) + half * 8;
        aF[t][e] = As[buf][wm * 32 + t * 16 + lm][ka];
        bF[t][e] = Bs[buf][wn * 32 + t * 16 + lm][ka];
      }
    }
    #pragma unroll
    for (int i = 0; i < 2; ++i)
      #pragma unroll
      for (int j = 0; j < 2; ++j)
        acc[i][j] = WMMA16(aF[i], bF[j], acc[i][j]);
    __syncthreads();                  // all waves done reading buf before refill
    if (kt + 2 < KT) issue(buf, kt + 2);
  }
  // C layout: lane 0-15 -> N=lane, rows v..; lanes 16-31 -> rows 8+v
  #pragma unroll
  for (int i = 0; i < 2; ++i)
    #pragma unroll
    for (int j = 0; j < 2; ++j)
      #pragma unroll
      for (int v = 0; v < 8; ++v){
        int row = m0 + wm * 32 + i * 16 + half * 8 + v;
        int col = n0 + wn * 32 + j * 16 + lm;
        size_t off = (size_t)row * N + col;
        float val = acc[i][j][v];
        if (EP == 0){ if (Res) val += Res[off]; C32[off] = val; }
        else if (EP == 1){ C16[off] = (f16)(0.5f * val * (1.0f + erff(val * 0.70710678118f))); }
        else { C16[off] = (f16)val; }
      }
}

// ---------------- flash attention, WMMA QK^T and P*V -----------------------
// One wave per (b, h, 16-query tile). DH=128, keys processed 32 at a time.
// Double-buffered async K/V staging; single wave32 per block means same-wave
// in-order LDS plus s_wait_asynccnt carry all the ordering (no barriers).
__global__ __launch_bounds__(32) void k_attn(const f16* __restrict__ qkv,
                                             f16* __restrict__ y16){
  const int qt = blockIdx.x;                 // 0..63
  const int bh = blockIdx.y;                 // 0..15
  const int b = bh >> 3, h = bh & 7;
  const int lane = threadIdx.x;
  const int half = lane >> 4, lm = lane & 15;
  const int q0 = qt * 16;
  const size_t rs = 3 * D_;                  // qkv row stride (halfs)
  const f16* base = qkv + (size_t)b * T_ * rs;
  __shared__ f16 Qs[16][128];
  __shared__ f16 Ks[2][32][128];
  __shared__ f16 Vs[2][32][128];
  __shared__ f16 Ps[16][40];
  auto issueKV = [&](int buf, int k0){
    #pragma unroll
    for (int i = 0; i < 16; ++i){
      int idx = lane + 32 * i;
      int r = idx >> 4, sc = idx & 15;
      async_copy16(&Ks[buf][r][sc * 8], base + (size_t)(k0 + r) * rs + D_     + h * 128 + sc * 8);
      async_copy16(&Vs[buf][r][sc * 8], base + (size_t)(k0 + r) * rs + 2 * D_ + h * 128 + sc * 8);
    }
  };
  #pragma unroll
  for (int i = 0; i < 8; ++i){               // async-stage Q (16x128 halfs)
    int idx = lane + 32 * i;
    int r = idx >> 4, sc = idx & 15;
    async_copy16(&Qs[r][sc * 8], base + (size_t)(q0 + r) * rs + h * 128 + sc * 8);
  }
  issueKV(0, 0);
  const v8f vz = {0.f,0.f,0.f,0.f,0.f,0.f,0.f,0.f};
  v8f out[8];
  #pragma unroll
  for (int dc = 0; dc < 8; ++dc) out[dc] = vz;
  float m_i[8], l_i[8];
  #pragma unroll
  for (int v = 0; v < 8; ++v){ m_i[v] = -3.0e38f; l_i[v] = 0.f; }
  const int ktmax = (q0 + 15) >> 5;
  for (int kt = 0; kt <= ktmax; ++kt){
    const int k0 = kt << 5;
    const int buf = kt & 1;
    if (kt + 1 <= ktmax){ issueKV(buf ^ 1, (kt + 1) << 5); wait_async32(); }
    else                { wait_async0(); }
    v8f s0 = vz, s1 = vz;                    // scores for keys [k0,k0+16) and [k0+16,k0+32)
    #pragma unroll
    for (int ks = 0; ks < 4; ++ks){
      v16h qa, kb0, kb1;
      #pragma unroll
      for (int e = 0; e < 16; ++e){
        int ka = (e < 8 ? e : e + 8) + half * 8;
        qa[e]  = Qs[lm][ks * 32 + ka];
        kb0[e] = Ks[buf][lm][ks * 32 + ka];
        kb1[e] = Ks[buf][16 + lm][ks * 32 + ka];
      }
      s0 = WMMA16(qa, kb0, s0);
      s1 = WMMA16(qa, kb1, s1);
    }
    const float sc = 0.08838834764831845f;   // 1/sqrt(128)
    float al[8];
    #pragma unroll
    for (int v = 0; v < 8; ++v){             // causal mask + online softmax per row
      int q = q0 + half * 8 + v;
      float a0 = (k0 + lm      <= q) ? s0[v] * sc : -1e30f;
      float a1 = (k0 + 16 + lm <= q) ? s1[v] * sc : -1e30f;
      float mv = fmaxf(a0, a1);
      #pragma unroll
      for (int off = 1; off < 16; off <<= 1) mv = fmaxf(mv, __shfl_xor(mv, off, 32));
      float mn = fmaxf(m_i[v], mv);
      float p0 = __expf(a0 - mn), p1 = __expf(a1 - mn);
      float rsum = p0 + p1;
      #pragma unroll
      for (int off = 1; off < 16; off <<= 1) rsum += __shfl_xor(rsum, off, 32);
      float ea = __expf(m_i[v] - mn);
      l_i[v] = l_i[v] * ea + rsum;
      m_i[v] = mn;
      al[v] = ea;
      Ps[half * 8 + v][lm]      = (f16)p0;   // stage P for A-fragment reshape
      Ps[half * 8 + v][16 + lm] = (f16)p1;
    }
    #pragma unroll
    for (int dc = 0; dc < 8; ++dc)
      #pragma unroll
      for (int v = 0; v < 8; ++v) out[dc][v] = out[dc][v] * al[v];
    // same-wave LDS is in-order: Ps stores above are seen by these loads
    v16h pa;
    #pragma unroll
    for (int e = 0; e < 16; ++e) pa[e] = Ps[lm][(e < 8 ? e : e + 8) + half * 8];
    #pragma unroll
    for (int dc = 0; dc < 8; ++dc){
      v16h vb;
      #pragma unroll
      for (int e = 0; e < 16; ++e) vb[e] = Vs[buf][(e < 8 ? e : e + 8) + half * 8][dc * 16 + lm];
      out[dc] = WMMA16(pa, vb, out[dc]);
    }
  }
  #pragma unroll
  for (int dc = 0; dc < 8; ++dc)
    #pragma unroll
    for (int v = 0; v < 8; ++v){
      int q = q0 + half * 8 + v;
      float val = out[dc][v] / l_i[v];
      y16[((size_t)(b * T_ + q)) * D_ + h * 128 + dc * 16 + lm] = (f16)val;
    }
}

// ---------------- host orchestration ----------------
static inline void launch_gemm(hipStream_t s, const f16* A, const f16* W, const float* Res,
                               float* C32, f16* C16, int M, int N, int K, int ep){
  dim3 g(N / 64, M / 128), blk(256);
  if (ep == 0)      k_gemm<0><<<g, blk, 0, s>>>(A, W, Res, C32, C16, M, N, K);
  else if (ep == 1) k_gemm<1><<<g, blk, 0, s>>>(A, W, Res, C32, C16, M, N, K);
  else              k_gemm<2><<<g, blk, 0, s>>>(A, W, Res, C32, C16, M, N, K);
}

extern "C" void kernel_launch(void* const* d_in, const int* in_sizes, int n_in,
                              void* d_out, int out_size, void* d_ws, size_t ws_size,
                              hipStream_t stream) {
  (void)in_sizes; (void)n_in; (void)out_size; (void)ws_size;
  const int*   ids      = (const int*)  d_in[0];
  const float* emb      = (const float*)d_in[1];
  const float* hy_n1    = (const float*)d_in[2];
  const float* hy_in    = (const float*)d_in[3];
  const float* hy_filt  = (const float*)d_in[4];
  const float* hy_o     = (const float*)d_in[5];
  const float* hy_n2    = (const float*)d_in[6];
  const float* hy_mi    = (const float*)d_in[7];
  const float* hy_mo    = (const float*)d_in[8];
  const float* bi_emb   = (const float*)d_in[9];
  const float* bi_proj  = (const float*)d_in[10];
  const float* tri_emb  = (const float*)d_in[11];
  const float* tri_proj = (const float*)d_in[12];
  const float* at_n1    = (const float*)d_in[13];
  const float* at_qkv   = (const float*)d_in[14];
  const float* at_o     = (const float*)d_in[15];
  const float* at_n2    = (const float*)d_in[16];
  const float* at_mi    = (const float*)d_in[17];
  const float* at_mo    = (const float*)d_in[18];
  const float* fn_w     = (const float*)d_in[19];
  float* logits = (float*)d_out;

  char* p = (char*)d_ws;
  auto carve = [&](size_t bytes) -> void* {
    void* r = p; p += (bytes + 255) & ~(size_t)255; return r;
  };
  float* x     = (float*)carve((size_t)2048 * 1024 * 4);   // residual stream f32
  f16*   h16   = (f16*)  carve((size_t)2048 * 1024 * 2);   // normed activations
  float* qkv32 = (float*)carve((size_t)2048 * 2048 * 4);   // hyena in-proj out
  f16*   sg16  = (f16*)  carve((size_t)2048 * 1024 * 2);   // conv*silu(gate)
  f16*   hid16 = (f16*)  carve((size_t)2048 * 4096 * 2);   // MLP hidden
  f16*   w16   = (f16*)  carve((size_t)4096 * 1024 * 2);   // weight staging
  f16*   qkv16 = (f16*)  carve((size_t)2048 * 3072 * 2);   // attention qkv
  f16*   y16   = (f16*)  carve((size_t)2048 * 1024 * 2);   // attention out
  f16*   emb16 = (f16*)  carve((size_t)32000 * 1024 * 2);  // vocab matrix f16

  k_embed<<<2048, 256, 0, stream>>>(ids, emb, x);

  for (int l = 0; l < 7; ++l){
    k_rmsnorm<<<2048, 256, 0, stream>>>(x, hy_n1 + (size_t)l * 1024, h16);
    k_fq<<<2048, 256, 0, stream>>>(hy_in + (size_t)l * 2048 * 1024, w16, 1024);
    launch_gemm(stream, h16, w16, nullptr, qkv32, nullptr, 2048, 2048, 1024, 0);
    k_scan<<<8, 256, 0, stream>>>(qkv32, hy_filt + (size_t)l * 1024 * 8, sg16);
    k_fq<<<1024, 256, 0, stream>>>(hy_o + (size_t)l * 1024 * 1024, w16, 1024);
    launch_gemm(stream, sg16, w16, x, x, nullptr, 2048, 1024, 1024, 0);
    k_rmsnorm<<<2048, 256, 0, stream>>>(x, hy_n2 + (size_t)l * 1024, h16);
    k_fq<<<4096, 256, 0, stream>>>(hy_mi + (size_t)l * 4096 * 1024, w16, 1024);
    launch_gemm(stream, h16, w16, nullptr, nullptr, hid16, 2048, 4096, 1024, 1);
    k_fq<<<1024, 256, 0, stream>>>(hy_mo + (size_t)l * 4096 * 1024, w16, 4096);
    launch_gemm(stream, hid16, w16, x, x, nullptr, 2048, 1024, 4096, 0);
  }

  k_hash<<<2048, 256, 0, stream>>>(ids, bi_emb, bi_proj, tri_emb, tri_proj, x);

  k_rmsnorm<<<2048, 256, 0, stream>>>(x, at_n1, h16);
  k_cvt<<<2048, 256, 0, stream>>>(at_qkv, w16, (size_t)3072 * 1024);
  launch_gemm(stream, h16, w16, nullptr, nullptr, qkv16, 2048, 3072, 1024, 2);
  k_attn<<<dim3(64, 16), 32, 0, stream>>>(qkv16, y16);
  k_cvt<<<1024, 256, 0, stream>>>(at_o, w16, (size_t)1024 * 1024);
  launch_gemm(stream, y16, w16, x, x, nullptr, 2048, 1024, 1024, 0);
  k_rmsnorm<<<2048, 256, 0, stream>>>(x, at_n2, h16);
  k_cvt<<<2048, 256, 0, stream>>>(at_mi, w16, (size_t)4096 * 1024);
  launch_gemm(stream, h16, w16, nullptr, nullptr, hid16, 2048, 4096, 1024, 1);
  k_cvt<<<2048, 256, 0, stream>>>(at_mo, w16, (size_t)4096 * 1024);
  launch_gemm(stream, hid16, w16, x, x, nullptr, 2048, 1024, 4096, 0);

  k_rmsnorm<<<2048, 256, 0, stream>>>(x, fn_w, h16);
  k_cvt<<<8192, 256, 0, stream>>>(emb, emb16, (size_t)32000 * 1024);
  launch_gemm(stream, h16, emb16, nullptr, logits, nullptr, 2048, 32000, 1024, 0);
}
